// KerasLMU_1400159338682
// MI455X (gfx1250) — matrix operand order repaired and checked by reading, not compiled, not run on hardware
//
#include <hip/hip_runtime.h>
#include <hip/hip_bf16.h>

// ---------------- problem constants ----------------
#define ORDER      256
#define HIDDEN     512
#define INPUT_DIM  256
#define BATCH      8192
#define TIMESTEPS  23
#define LEAKY      0.2f

#define BT   32          // batch rows per workgroup (2 row-blocks of 16)
#define NTHR 256         // 8 waves of 32

typedef __attribute__((ext_vector_type(16))) __bf16 bf16x16;
typedef __attribute__((ext_vector_type(8)))  __bf16 bf16x8;
typedef __attribute__((ext_vector_type(8)))  float  f32x8;

// ---------------- WMMA helper ----------------
__device__ __forceinline__ f32x8 wmma_bf16(bf16x16 a, bf16x16 b, f32x8 c) {
  // (neg_a, A, neg_b, B, c_mod, C, reuse_a, reuse_b)
  return __builtin_amdgcn_wmma_f32_16x16x32_bf16(false, a, false, b, (short)0, c,
                                                 false, false);
}

// A-operand (16x32 bf16) from row-major bf16 LDS.
// lane<16 : row M = lane,    K chunks {0..15}
// lane>=16: row M = lane-16, K chunks {16..31}
__device__ __forceinline__ bf16x16 load_a(const __bf16* rowp, int kb, int laneHi) {
  const __bf16* p = rowp + kb * 32 + laneHi * 16;
  bf16x8 lo = *(const bf16x8*)(p);
  bf16x8 hi = *(const bf16x8*)(p + 8);
  return __builtin_shufflevector(lo, hi, 0,1,2,3,4,5,6,7,8,9,10,11,12,13,14,15);
}

// B-operand (32x16 bf16): per-lane base pointer (already includes lane*16
// elements and the wave's tile offset) + compile-time-constant element offset
// so loads fold into base + 24-bit instruction offset immediates.
__device__ __forceinline__ bf16x16 load_b(const __bf16* p, int coff) {
  bf16x8 lo = *(const bf16x8*)(p + coff);
  bf16x8 hi = *(const bf16x8*)(p + coff + 8);
  return __builtin_shufflevector(lo, hi, 0,1,2,3,4,5,6,7,8,9,10,11,12,13,14,15);
}

// ---------------- prologue: u[b][t] = x_t . e_x ----------------
__global__ void lmu_u_kernel(const float* __restrict__ inputs,
                             const float* __restrict__ e_x,
                             float* __restrict__ U) {
  int idx = blockIdx.x * blockDim.x + threadIdx.x;       // idx = b*T + t
  if (idx >= BATCH * TIMESTEPS) return;
  const float* x = inputs + (size_t)idx * INPUT_DIM;     // (B,T,K) contiguous
  float s = 0.f;
  for (int k = 0; k < INPUT_DIM; ++k) s += x[k] * e_x[k];
  U[idx] = s;
}

// ---------------- prologue: pack W[k][n] = src[k*sk + n*sn] into bf16
// B-operand tiles. tile = nb*Kb + kb; lane holds column n = nb*16+(lane&15),
// K = kb*32 + (lane>>4)*16 + e, e = 0..15.
__global__ void lmu_pack_kernel(const float* __restrict__ src,
                                __bf16* __restrict__ dst,
                                int Kb, int sk, int sn, int nTiles) {
  int tid  = blockIdx.x * blockDim.x + threadIdx.x;
  int tile = tid >> 5;
  if (tile >= nTiles) return;
  int lane  = tid & 31;
  int nb    = tile / Kb;
  int kb    = tile % Kb;
  int n     = nb * 16 + (lane & 15);
  int kbase = kb * 32 + (lane >> 4) * 16;
  __bf16* d = dst + (size_t)tile * 512 + lane * 16;
#pragma unroll
  for (int e = 0; e < 16; ++e)
    d[e] = (__bf16)src[(size_t)(kbase + e) * sk + (size_t)n * sn];
}

// ---------------- main recurrent kernel ----------------
__global__ __launch_bounds__(NTHR) void lmu_main(
    const float*  __restrict__ inputs,  // [B][T][256] f32
    const float*  __restrict__ Bv,      // [256] f32
    const float*  __restrict__ U,       // [B][T] f32
    const __bf16* __restrict__ packA,   // 16 nb x  8 kb tiles
    const __bf16* __restrict__ packWx,  // 32 nb x  8 kb tiles
    const __bf16* __restrict__ packWh,  // 32 nb x 16 kb tiles
    const __bf16* __restrict__ packWm,  // 32 nb x  8 kb tiles
    float* __restrict__ out)            // [B][T][512] f32
{
  __shared__ __bf16 sH[BT * HIDDEN];    // 32 KB
  __shared__ __bf16 sM[BT * ORDER];     // 16 KB
  __shared__ __bf16 sX[BT * INPUT_DIM]; // 16 KB  (total exactly 64 KB)

  const int tid    = threadIdx.x;
  const int wave   = tid >> 5;
  const int lane   = tid & 31;
  const int laneHi = lane >> 4;     // 0 / 1
  const int laneN  = lane & 15;     // column within 16x16 tile
  const int bBase  = blockIdx.x * BT;
  const int laneB  = lane * 16;     // element offset of this lane inside a B tile

  for (int i = tid; i < BT * HIDDEN; i += NTHR) sH[i] = (__bf16)0.f;
  for (int i = tid; i < BT * ORDER;  i += NTHR) sM[i] = (__bf16)0.f;
  __syncthreads();

  // Opaque zero, redefined by a volatile asm each timestep: weight base
  // pointers become loop-variant (no LICM across t -> no scratch spills)
  // while remaining visible to address-space inference (global, not flat).
  int zero = 0;

#pragma unroll 1
  for (int t = 0; t < TIMESTEPS; ++t) {
    asm volatile("" : "+s"(zero));

    // Per-phase, per-lane base pointers: all wave/lane-dependent parts are
    // folded in once; inner-loop offsets are compile-time constants that
    // become instruction offset immediates.
    const __bf16* pA  = packA  + zero + (wave * 2) * (8  * 512) + laneB;
    const __bf16* pWx = packWx + zero + (wave * 4) * (8  * 512) + laneB;
    const __bf16* pWh = packWh + zero + (wave * 4) * (16 * 512) + laneB;
    const __bf16* pWm = packWm + zero + (wave * 4) * (8  * 512) + laneB;

    // ---- stage x_t as bf16 in LDS (coalesced f32 reads) ----
    for (int i = tid; i < BT * INPUT_DIM; i += NTHR) {
      int r = i >> 8, k = i & 255;
      sX[i] = (__bf16)inputs[(size_t)(bBase + r) * (TIMESTEPS * INPUT_DIM)
                             + t * INPUT_DIM + k];
    }
    __syncthreads();

    // ---- Phase A: m_new = m @ A^T ----
    // Wave owns nb = wave*2 + j (j=0..1), both row-blocks. Per K-block:
    // 2 A ds-loads, 2 B global loads, 4 WMMAs (A and B each reused 2x).
    f32x8 accA[2][2];  // [j][rb]
#pragma unroll
    for (int j = 0; j < 2; ++j)
#pragma unroll
      for (int rb = 0; rb < 2; ++rb) { f32x8 z = {}; accA[j][rb] = z; }
    {
      const __bf16* mrow0 = sM + (0  + laneN) * ORDER;
      const __bf16* mrow1 = sM + (16 + laneN) * ORDER;
#pragma unroll
      for (int kb = 0; kb < 8; ++kb) {
        bf16x16 a0 = load_a(mrow0, kb, laneHi);
        bf16x16 a1 = load_a(mrow1, kb, laneHi);
#pragma unroll
        for (int j = 0; j < 2; ++j) {
          bf16x16 b = load_b(pA, (j * 8 + kb) * 512);
          accA[j][0] = wmma_bf16(a0, b, accA[j][0]);
          accA[j][1] = wmma_bf16(a1, b, accA[j][1]);
        }
      }
    }
    __syncthreads();                 // all waves done reading old m

    // ---- write m_new = acc + u*Bv back to sM (bf16) ----
#pragma unroll
    for (int j = 0; j < 2; ++j) {
      int col = (wave * 2 + j) * 16 + laneN;
      float bvv = Bv[col];
#pragma unroll
      for (int rb = 0; rb < 2; ++rb)
#pragma unroll
        for (int v = 0; v < 8; ++v) {
          int row = rb * 16 + v + laneHi * 8;      // C/D layout row
          float uval = U[(size_t)(bBase + row) * TIMESTEPS + t];
          sM[row * ORDER + col] = (__bf16)(accA[j][rb][v] + uval * bvv);
        }
    }
    __syncthreads();

    // ---- Phase B: pre = x@Wx + h@Wh^T + m_new@Wm^T ----
    // Wave owns nb = wave*4 + j (j=0..3), both row-blocks: 8 accumulators.
    f32x8 accB[4][2];  // [j][rb]
#pragma unroll
    for (int j = 0; j < 4; ++j)
#pragma unroll
      for (int rb = 0; rb < 2; ++rb) { f32x8 z = {}; accB[j][rb] = z; }

    const __bf16* xrow0 = sX + (0  + laneN) * INPUT_DIM;
    const __bf16* xrow1 = sX + (16 + laneN) * INPUT_DIM;
    const __bf16* hrow0 = sH + (0  + laneN) * HIDDEN;
    const __bf16* hrow1 = sH + (16 + laneN) * HIDDEN;
    const __bf16* mrow0 = sM + (0  + laneN) * ORDER;
    const __bf16* mrow1 = sM + (16 + laneN) * ORDER;

#pragma unroll
    for (int kb = 0; kb < 8; ++kb) {            // x @ Wx  (K = 256)
      bf16x16 a0 = load_a(xrow0, kb, laneHi);
      bf16x16 a1 = load_a(xrow1, kb, laneHi);
#pragma unroll
      for (int j = 0; j < 4; ++j) {
        bf16x16 b = load_b(pWx, (j * 8 + kb) * 512);
        accB[j][0] = wmma_bf16(a0, b, accB[j][0]);
        accB[j][1] = wmma_bf16(a1, b, accB[j][1]);
      }
    }
#pragma unroll
    for (int kb = 0; kb < 16; ++kb) {           // h @ Wh^T (K = 512)
      bf16x16 a0 = load_a(hrow0, kb, laneHi);
      bf16x16 a1 = load_a(hrow1, kb, laneHi);
#pragma unroll
      for (int j = 0; j < 4; ++j) {
        bf16x16 b = load_b(pWh, (j * 16 + kb) * 512);
        accB[j][0] = wmma_bf16(a0, b, accB[j][0]);
        accB[j][1] = wmma_bf16(a1, b, accB[j][1]);
      }
    }
#pragma unroll
    for (int kb = 0; kb < 8; ++kb) {            // m_new @ Wm^T (K = 256)
      bf16x16 a0 = load_a(mrow0, kb, laneHi);
      bf16x16 a1 = load_a(mrow1, kb, laneHi);
#pragma unroll
      for (int j = 0; j < 4; ++j) {
        bf16x16 b = load_b(pWm, (j * 8 + kb) * 512);
        accB[j][0] = wmma_bf16(a0, b, accB[j][0]);
        accB[j][1] = wmma_bf16(a1, b, accB[j][1]);
      }
    }
    __syncthreads();                 // all waves done reading old h

    // ---- leaky ReLU, write h_new to LDS (bf16) and d_out (f32) ----
#pragma unroll
    for (int j = 0; j < 4; ++j) {
      int col = (wave * 4 + j) * 16 + laneN;
#pragma unroll
      for (int rb = 0; rb < 2; ++rb)
#pragma unroll
        for (int v = 0; v < 8; ++v) {
          int row = rb * 16 + v + laneHi * 8;
          float p  = accB[j][rb][v];
          float hv = p >= 0.f ? p : LEAKY * p;
          sH[row * HIDDEN + col] = (__bf16)hv;
          out[(size_t)(bBase + row) * (TIMESTEPS * HIDDEN)
              + (size_t)t * HIDDEN + col] = hv;
        }
    }
    __syncthreads();
  }
}

// ---------------- launch ----------------
extern "C" void kernel_launch(void* const* d_in, const int* in_sizes, int n_in,
                              void* d_out, int out_size, void* d_ws, size_t ws_size,
                              hipStream_t stream) {
  // setup_inputs order: inputs, A, B, W_h, W_m, W_x, e_x   (all f32)
  const float* inputs = (const float*)d_in[0];
  const float* A      = (const float*)d_in[1];
  const float* Bm     = (const float*)d_in[2];   // (256,1) -> Bv
  const float* W_h    = (const float*)d_in[3];   // (512,512)
  const float* W_m    = (const float*)d_in[4];   // (512,256)
  const float* W_x    = (const float*)d_in[5];   // (256,512)
  const float* e_x    = (const float*)d_in[6];   // (256,1)
  float*       out    = (float*)d_out;

  char* ws = (char*)d_ws;
  float*  U      = (float*) (ws + 0);            // 8192*23*4      = 753664
  __bf16* packA  = (__bf16*)(ws + 753664);       // 128 tiles*1KB  = 131072
  __bf16* packWx = (__bf16*)(ws + 884736);       // 256 tiles      = 262144
  __bf16* packWh = (__bf16*)(ws + 1146880);      // 512 tiles      = 524288
  __bf16* packWm = (__bf16*)(ws + 1671168);      // 256 tiles      = 262144
  (void)in_sizes; (void)n_in; (void)out_size; (void)ws_size;

  lmu_u_kernel<<<(BATCH * TIMESTEPS + 255) / 256, 256, 0, stream>>>(inputs, e_x, U);

  // m @ A^T  : W[k][n] = A[n][k]    K=256 N=256
  lmu_pack_kernel<<<(128 * 32) / 256, 256, 0, stream>>>(A,   packA,   8, 1,   256, 128);
  // x @ Wx   : W[k][n] = Wx[k][n]   K=256 N=512
  lmu_pack_kernel<<<(256 * 32) / 256, 256, 0, stream>>>(W_x, packWx,  8, 512, 1,   256);
  // h @ Wh^T : W[k][n] = Wh[n][k]   K=512 N=512
  lmu_pack_kernel<<<(512 * 32) / 256, 256, 0, stream>>>(W_h, packWh, 16, 1,   512, 512);
  // m @ Wm^T : W[k][n] = Wm[n][k]   K=256 N=512
  lmu_pack_kernel<<<(256 * 32) / 256, 256, 0, stream>>>(W_m, packWm,  8, 1,   256, 256);

  lmu_main<<<BATCH / BT, NTHR, 0, stream>>>(inputs, Bm, U, packA, packWx, packWh,
                                            packWm, out);
}
